// InteractionSitesModule_75136157877009
// MI455X (gfx1250) — matrix operand order repaired and checked by reading, not compiled, not run on hardware
//
#include <hip/hip_runtime.h>
#include <hip/hip_bf16.h>
#include <math.h>

typedef __attribute__((ext_vector_type(16))) _Float16 v16h;
typedef __attribute__((ext_vector_type(8)))  _Float16 v8h;
typedef __attribute__((ext_vector_type(8)))  float    v8f;

#define HID 768
#define PROJ 64
#define NB 32
#define LC 128
#define LP 1024

// ---- workspace layout (bytes) ----
#define OFF_PWC   0u                      // f16 [4096,64]   = 512 KB
#define OFF_PWP   524288u                 // f16 [32768,64]  = 4 MB
#define OFF_LATC  4718592u                // f32 [4096,64]   = 1 MB
#define OFF_LATP  5767168u                // f32 [32768,64]  = 8 MB
#define OFF_ROW   14155776u               // f32 [4096]
#define OFF_COL   14172160u               // f32 [32768]
#define OFF_WT    14303232u               // f16 [4][64][768] = 384 KB (transposed weights)
// total ~14.7 MB

__device__ __forceinline__ float gelu_exact(float x) {
    return 0.5f * x * (1.0f + erff(x * 0.70710678118654752440f));
}

// ------------------------------------------------------------------
// zero row/col accumulators
// ------------------------------------------------------------------
__global__ void zero_acc_kernel(float* __restrict__ p, int n) {
    int i = blockIdx.x * blockDim.x + threadIdx.x;
    if (i < n) p[i] = 0.0f;
}

// ------------------------------------------------------------------
// One-time weight prep: Wt[w][n][k] = (f16) W_w[k][n]
// Puts each output-column's K-vector contiguous so the WMMA B-fragment
// is a single 32-byte vector load per lane with no conversion in the
// GEMM hot loop.  w order: 0=pc, 1=pp, 2=lc, 3=lp.
// ------------------------------------------------------------------
__global__ void wt_kernel(const float* __restrict__ W_pc, const float* __restrict__ W_pp,
                          const float* __restrict__ W_lc, const float* __restrict__ W_lp,
                          _Float16* __restrict__ Wt) {
    int idx = blockIdx.x * blockDim.x + threadIdx.x;       // < 4*64*768
    if (idx >= 4 * PROJ * HID) return;
    const int w = idx / (PROJ * HID);
    const int r = idx - w * (PROJ * HID);
    const int n = r / HID;
    const int k = r - n * HID;
    const float* W = (w == 0) ? W_pc : (w == 1) ? W_pp : (w == 2) ? W_lc : W_lp;
    Wt[idx] = (_Float16)W[(size_t)k * PROJ + n];
}

// ------------------------------------------------------------------
// Projection: out = GELU(X @ W + b) * mask
// One wave owns 16 rows and computes the FULL 64-wide output for BOTH
// weight matrices of its input (8 accumulators = 2 wsel x 4 ntile).
// A fragment: two 32B float4-pairs per lane, converted once per K-step,
// reused by 8 WMMAs.  B fragment: one 32B v16h load (pre-transposed f16).
// ------------------------------------------------------------------
__global__ __launch_bounds__(128)
void proj_kernel(const float* __restrict__ protein,   // [32768,768]
                 const float* __restrict__ pmask,     // [32768]
                 const float* __restrict__ compound,  // [4096,768]
                 const float* __restrict__ cmask,     // [4096]
                 const _Float16* __restrict__ Wt,     // [4][64][768] f16
                 const float* __restrict__ b_pc, const float* __restrict__ b_pp,
                 const float* __restrict__ b_lc, const float* __restrict__ b_lp,
                 _Float16* __restrict__ pwc, _Float16* __restrict__ pwp,
                 float* __restrict__ latc, float* __restrict__ latp) {
    const int lane = threadIdx.x & 31;
    const int tile = blockIdx.x * 4 + (threadIdx.x >> 5);   // < 2304 row tiles

    const bool isP = (tile >= 256);
    const int rowtile = isP ? (tile - 256) : tile;
    const int m0 = rowtile << 4;

    const float* X    = isP ? protein : compound;
    const float* mask = isP ? pmask   : cmask;
    const _Float16* Wtsel[2] = { Wt + (size_t)(isP ? 1 : 0) * PROJ * HID,    // pw weights
                                 Wt + (size_t)(isP ? 3 : 2) * PROJ * HID };  // lat weights
    const float* Bsel[2]     = { isP ? b_pp : b_pc, isP ? b_lp : b_lc };
    _Float16* pwout  = isP ? pwp  : pwc;
    float*    latout = isP ? latp : latc;

    // A-fragment (16-bit A, 16x32 MxK): lane holds m = lane&15;
    //  halves 0..7  = K (lane<16 ? 0..7  : 16..23)
    //  halves 8..15 = K (lane<16 ? 8..15 : 24..31)
    const int am  = lane & 15;
    const int akb = (lane >> 4) * 16;
    const float* xrow = X + (size_t)(m0 + am) * HID;

    // B-fragment (16-bit B, 32x16 KxN): lane n = lane&15 holds K
    //  (lane<16 ? 0..15 : 16..31) contiguous in transposed Wt.
    const int bn  = lane & 15;
    const int bkh = (lane >> 4) * 16;

    v8f acc[8] = {};                                        // [wsel][ntile]

    for (int k0 = 0; k0 < HID; k0 += 32) {
        const float4 lo0 = *reinterpret_cast<const float4*>(xrow + k0 + akb);
        const float4 lo1 = *reinterpret_cast<const float4*>(xrow + k0 + akb + 4);
        const float4 hi0 = *reinterpret_cast<const float4*>(xrow + k0 + 8 + akb);
        const float4 hi1 = *reinterpret_cast<const float4*>(xrow + k0 + 8 + akb + 4);
        v16h a;
        a[0]  = (_Float16)lo0.x;  a[1]  = (_Float16)lo0.y;
        a[2]  = (_Float16)lo0.z;  a[3]  = (_Float16)lo0.w;
        a[4]  = (_Float16)lo1.x;  a[5]  = (_Float16)lo1.y;
        a[6]  = (_Float16)lo1.z;  a[7]  = (_Float16)lo1.w;
        a[8]  = (_Float16)hi0.x;  a[9]  = (_Float16)hi0.y;
        a[10] = (_Float16)hi0.z;  a[11] = (_Float16)hi0.w;
        a[12] = (_Float16)hi1.x;  a[13] = (_Float16)hi1.y;
        a[14] = (_Float16)hi1.z;  a[15] = (_Float16)hi1.w;

#pragma unroll
        for (int w = 0; w < 2; ++w) {
#pragma unroll
            for (int nt = 0; nt < 4; ++nt) {
                const v16h b = *reinterpret_cast<const v16h*>(
                    Wtsel[w] + (size_t)((nt << 4) + bn) * HID + k0 + bkh);
                acc[w * 4 + nt] = __builtin_amdgcn_wmma_f32_16x16x32_f16(
                    false, a, false, b, (short)0, acc[w * 4 + nt], false, false);
            }
        }
    }

    // Epilogue: C/D layout -> lane n = lane&15, m = (lane>>4)*8 + v
    const int cn  = lane & 15;
    const int cmh = (lane >> 4) * 8;
    float mrow[8];
#pragma unroll
    for (int v = 0; v < 8; ++v) mrow[v] = mask[m0 + cmh + v];

#pragma unroll
    for (int w = 0; w < 2; ++w) {
        const float* bias = Bsel[w];
#pragma unroll
        for (int nt = 0; nt < 4; ++nt) {
            const int n = (nt << 4) + cn;
            const float bv = bias[n];
#pragma unroll
            for (int v = 0; v < 8; ++v) {
                const int m = m0 + cmh + v;
                const float g = gelu_exact(acc[w * 4 + nt][v] + bv) * mrow[v];
                const size_t idx = (size_t)m * PROJ + n;
                if (w) latout[idx] = g;
                else   pwout[idx]  = (_Float16)g;
            }
        }
    }
}

// ------------------------------------------------------------------
// Pairwise: map[b,i,j] = sum_f pw_c[b,i,f]*pw_p[b,j,f]   (K = 64)
// One wave per 16x16 tile. Epilogue fuses:
//   d_out tile  = sigmoid(map) * cmask[i]*pmask[j]
//   row[b,i]   += sum_j map*pairwise_mask   (cross-lane shuffle + atomics)
//   col[b,j]   += sum_i map*pairwise_mask
// ------------------------------------------------------------------
__global__ __launch_bounds__(128)
void pair_kernel(const _Float16* __restrict__ pwc,   // [4096,64]
                 const _Float16* __restrict__ pwp,   // [32768,64]
                 const float* __restrict__ pairmask, // [32,128,1024]
                 const float* __restrict__ cmask,    // [4096]
                 const float* __restrict__ pmask,    // [32768]
                 float* __restrict__ out_pair,       // d_out + 4096
                 float* __restrict__ rowAcc,         // [4096]
                 float* __restrict__ colAcc) {       // [32768]
    const int lane = threadIdx.x & 31;
    int tile = blockIdx.x * 4 + (threadIdx.x >> 5);   // < 16384
    const int j0 = (tile & 63) << 4;  tile >>= 6;
    const int i0 = (tile & 7)  << 4;  tile >>= 3;
    const int b  = tile;                              // < 32

    const int am  = lane & 15;
    const int akb = (lane >> 4) * 16;
    const _Float16* arow = pwc + (size_t)(b * LC + i0 + am) * PROJ;

    const int bn  = lane & 15;
    const int bkh = (lane >> 4) * 16;
    const _Float16* brow = pwp + (size_t)(b * LP + j0 + bn) * PROJ;

    v8f acc = {};
#pragma unroll
    for (int k0 = 0; k0 < PROJ; k0 += 32) {
        const v8h alo = *reinterpret_cast<const v8h*>(arow + k0 + akb);
        const v8h ahi = *reinterpret_cast<const v8h*>(arow + k0 + 8 + akb);
        const v16h a = __builtin_shufflevector(alo, ahi,
            0, 1, 2, 3, 4, 5, 6, 7, 8, 9, 10, 11, 12, 13, 14, 15);
        const v16h bb = *reinterpret_cast<const v16h*>(brow + k0 + bkh);
        acc = __builtin_amdgcn_wmma_f32_16x16x32_f16(false, a, false, bb,
                                                     (short)0, acc, false, false);
    }

    const int cn  = lane & 15;
    const int cmh = (lane >> 4) * 8;
    const int j   = j0 + cn;
    const float pmj = pmask[b * LP + j];

    float P[8];
    float colsum = 0.0f;
#pragma unroll
    for (int v = 0; v < 8; ++v) {
        const int i = i0 + cmh + v;
        const float map = acc[v];
        const size_t oidx = ((size_t)(b * LC + i)) * LP + j;
        const float sg = 1.0f / (1.0f + __expf(-map));
        out_pair[oidx] = sg * cmask[b * LC + i] * pmj;
        P[v] = map * pairmask[oidx];
        colsum += P[v];
    }

    // column partial sums: add the two lane halves, lanes 0-15 commit
    colsum += __shfl_xor(colsum, 16, 32);
    if (lane < 16) atomicAdd(&colAcc[b * LP + j], colsum);

    // row partial sums: reduce over the 16 lanes of each half
    float rs[8];
#pragma unroll
    for (int v = 0; v < 8; ++v) {
        float r = P[v];
        r += __shfl_xor(r, 1, 32);
        r += __shfl_xor(r, 2, 32);
        r += __shfl_xor(r, 4, 32);
        r += __shfl_xor(r, 8, 32);
        rs[v] = r;
    }
    if ((lane & 15) == 0) {
        const int ib = b * LC + i0 + cmh;
#pragma unroll
        for (int v = 0; v < 8; ++v) atomicAdd(&rowAcc[ib + v], rs[v]);
    }
}

// ------------------------------------------------------------------
// latent_vectors[b, 0:64]  = sum_i row[b,i] * lat_c[b,i,:]
// latent_vectors[b,64:128] = sum_j col[b,j] * lat_p[b,j,:]
// ------------------------------------------------------------------
__global__ __launch_bounds__(128)
void latent_kernel(const float* __restrict__ rowAcc, const float* __restrict__ colAcc,
                   const float* __restrict__ latc, const float* __restrict__ latp,
                   float* __restrict__ out_latent) {
    const int b = blockIdx.x;
    const int t = threadIdx.x;
    if (t < 64) {
        const int f = t;
        float s = 0.0f;
        for (int i = 0; i < LC; ++i)
            s += rowAcc[b * LC + i] * latc[(size_t)(b * LC + i) * PROJ + f];
        out_latent[b * 128 + f] = s;
    } else {
        const int f = t - 64;
        float s = 0.0f;
        for (int j = 0; j < LP; ++j)
            s += colAcc[b * LP + j] * latp[(size_t)(b * LP + j) * PROJ + f];
        out_latent[b * 128 + 64 + f] = s;
    }
}

extern "C" void kernel_launch(void* const* d_in, const int* in_sizes, int n_in,
                              void* d_out, int out_size, void* d_ws, size_t ws_size,
                              hipStream_t stream) {
    const float* protein  = (const float*)d_in[0];
    const float* pmask    = (const float*)d_in[1];
    const float* compound = (const float*)d_in[2];
    const float* cmask    = (const float*)d_in[3];
    const float* pairmask = (const float*)d_in[4];
    const float* W_pc = (const float*)d_in[5];  const float* b_pc = (const float*)d_in[6];
    const float* W_pp = (const float*)d_in[7];  const float* b_pp = (const float*)d_in[8];
    const float* W_lc = (const float*)d_in[9];  const float* b_lc = (const float*)d_in[10];
    const float* W_lp = (const float*)d_in[11]; const float* b_lp = (const float*)d_in[12];

    char* ws = (char*)d_ws;
    _Float16* pwc  = (_Float16*)(ws + OFF_PWC);
    _Float16* pwp  = (_Float16*)(ws + OFF_PWP);
    float*    latc = (float*)(ws + OFF_LATC);
    float*    latp = (float*)(ws + OFF_LATP);
    float*    rowA = (float*)(ws + OFF_ROW);
    float*    colA = (float*)(ws + OFF_COL);
    _Float16* Wt   = (_Float16*)(ws + OFF_WT);

    float* out   = (float*)d_out;            // [32,128] latent first
    float* out_p = out + NB * 128;           // then [32,128,1024]

    // 1) clear row/col accumulators (4096 + 32768 floats; contiguous in ws)
    zero_acc_kernel<<<(NB * (LC + LP) + 255) / 256, 256, 0, stream>>>(rowA, NB * (LC + LP));

    // 2) transpose + f16-convert weights (one-time prep, 196K elems)
    wt_kernel<<<(4 * PROJ * HID + 255) / 256, 256, 0, stream>>>(W_pc, W_pp, W_lc, W_lp, Wt);

    // 3) projections: 2304 row tiles, 4 waves/block, 8 WMMA accumulators/wave
    proj_kernel<<<2304 / 4, 128, 0, stream>>>(protein, pmask, compound, cmask,
                                              Wt, b_pc, b_pp, b_lc, b_lp,
                                              pwc, pwp, latc, latp);

    // 4) pairwise GEMM + fused epilogue: 16384 tiles, 4 waves/block
    pair_kernel<<<16384 / 4, 128, 0, stream>>>(pwc, pwp, pairmask, cmask, pmask,
                                               out_p, rowA, colA);

    // 5) latent vectors
    latent_kernel<<<NB, 128, 0, stream>>>(rowA, colA, latc, latp, out);
}
